// GNNSDFFixedK_21912923144200
// MI455X (gfx1250) — matrix operand order repaired and checked by reading, not compiled, not run on hardware
//
#include <hip/hip_runtime.h>

typedef __attribute__((ext_vector_type(16))) _Float16 v16h;
typedef __attribute__((ext_vector_type(8)))  _Float16 v8h;
typedef __attribute__((ext_vector_type(8)))  float    v8f;

// ---- f16 transposed-weight workspace layout (offsets in halfs) ----
#define OFF_L1_W1T 0      // [32][32]  (in 8  -> pad 32)
#define OFF_L1_W2T 1024   // [16][32]
#define OFF_L2_W1T 1536   // [32][32]  (in 24 -> pad 32)
#define OFF_L2_W2T 2560   // [16][32]
#define OFF_L3_W1T 3072   // [32][32]
#define OFF_L3_W2T 4096   // [16][32]
#define OFF_L4_W1T 4608   // [64][32]  (in 27 -> pad 32)
#define OFF_L4_W2T 6656   // [16][64]  (out 1 -> pad 16 rows)
#define WTS_TOTAL  7680

// Convert f32 weights -> f16, transposed [N][K] with zero padding.
__global__ void prep_weights(const float* w1a, const float* w2a,
                             const float* w1b, const float* w2b,
                             const float* w1c, const float* w2c,
                             const float* w1d, const float* w2d,
                             _Float16* ws) {
    int idx = blockIdx.x * blockDim.x + threadIdx.x;
    if (idx >= WTS_TOTAL) return;
    float v;
    if (idx < 1536) {
        if (idx < 1024) { int n = idx >> 5, k = idx & 31; v = (k < 8)  ? w1a[k*32 + n] : 0.f; }
        else            { int t = idx-1024; int n = t >> 5, k = t & 31; v = w2a[k*16 + n]; }
    } else if (idx < 3072) {
        if (idx < 2560) { int t = idx-1536; int n = t >> 5, k = t & 31; v = (k < 24) ? w1b[k*32 + n] : 0.f; }
        else            { int t = idx-2560; int n = t >> 5, k = t & 31; v = w2b[k*16 + n]; }
    } else if (idx < 4608) {
        if (idx < 4096) { int t = idx-3072; int n = t >> 5, k = t & 31; v = (k < 24) ? w1c[k*32 + n] : 0.f; }
        else            { int t = idx-4096; int n = t >> 5, k = t & 31; v = w2c[k*16 + n]; }
    } else if (idx < 6656) {
        int t = idx-4608; int n = t >> 5, k = t & 31; v = (k < 27) ? w1d[k*64 + n] : 0.f;
    } else {
        int t = idx-6656; int n = t >> 6, k = t & 63; v = (n == 0) ? w2d[k] : 0.f; // W2 l4 is (64,1)
    }
    ws[idx] = (_Float16)v;
}

// ---- WMMA fragment helpers (wave32, 16x16x32 f16) ----

// A: 16x32 f16 tile from LDS row-major [row][K] (stride halfs).
// lanes 0-15 hold M=0..15 with K {0..7, 16..23}; lanes 16-31 same M, K {8..15, 24..31}.
__device__ inline v16h afrag(const _Float16* buf, int stride, int row0, int lane, int kbase) {
    int row  = row0 + (lane & 15);
    int koff = (lane >> 4) << 3;                 // 0 or 8
    const _Float16* p = buf + row * stride + kbase + koff;
    v8h lo = *(const v8h*)(p);
    v8h hi = *(const v8h*)(p + 16);
    v16h a;
#pragma unroll
    for (int i = 0; i < 8; ++i) { a[i] = lo[i]; a[8 + i] = hi[i]; }
    return a;
}

// B: 32x16 f16 tile from global transposed weights [N][K] (ldk halfs).
// lane holds column N=n0+(lane&15); K {0..7, 16..23} (lanes 0-15) or K {8..15, 24..31} (lanes 16-31).
__device__ inline v16h bfrag(const _Float16* wT, int ldk, int n0, int lane, int kbase) {
    int n  = n0 + (lane & 15);
    int kb = kbase + ((lane >> 4) << 3);         // 0 or 8 (+kbase)
    const _Float16* p = wT + n * ldk + kb;
    v8h lo = *(const v8h*)(p);
    v8h hi = *(const v8h*)(p + 16);
    v16h b;
#pragma unroll
    for (int i = 0; i < 8; ++i) { b[i] = lo[i]; b[8 + i] = hi[i]; }
    return b;
}

// C init: broadcast bias per output column (C layout: N = lane&15 across lanes, M across VGPRs).
__device__ inline v8f cinit(const float* b, int n0, int lane) {
    float bv = b[n0 + (lane & 15)];
    v8f c;
#pragma unroll
    for (int r = 0; r < 8; ++r) c[r] = bv;
    return c;
}

// Store C/D tile to LDS f16 row-major [M][col] (stride halfs). VGPR r: M = r (lanes<16) or 8+r.
// ReLU via fmaxf -> single v_max_num_f32 (no cmp+select).
__device__ inline void store_c(_Float16* buf, int stride, int row0, int n0, int lane, v8f c, bool relu) {
    int n  = n0 + (lane & 15);
    int mb = row0 + ((lane >> 4) << 3);
#pragma unroll
    for (int r = 0; r < 8; ++r) {
        float v = c[r];
        if (relu) v = fmaxf(v, 0.f);
        buf[(mb + r) * stride + n] = (_Float16)v;
    }
}

#define WMMA(a, b, c) __builtin_amdgcn_wmma_f32_16x16x32_f16(false, (a), false, (b), (short)0, (c), false, false)

// Per-edge MLP: K=32 in (abuf [32][32]) -> H=32 relu (hbuf) -> O=16 back into abuf cols 0..15.
__device__ inline void edge_mlp(_Float16* ab, _Float16* hb,
                                const _Float16* w1T, const float* b1,
                                const _Float16* w2T, const float* b2, int lane) {
    __syncthreads();
#pragma unroll
    for (int mt = 0; mt < 2; ++mt) {
        v16h a = afrag(ab, 32, mt * 16, lane, 0);
#pragma unroll
        for (int nt = 0; nt < 2; ++nt) {
            v16h b = bfrag(w1T, 32, nt * 16, lane, 0);
            v8f  c = cinit(b1, nt * 16, lane);
            c = WMMA(a, b, c);
            store_c(hb, 64, mt * 16, nt * 16, lane, c, true);
        }
    }
    __syncthreads();
#pragma unroll
    for (int mt = 0; mt < 2; ++mt) {
        v16h a = afrag(hb, 64, mt * 16, lane, 0);
        v16h b = bfrag(w2T, 32, 0, lane, 0);
        v8f  c = cinit(b2, 0, lane);
        c = WMMA(a, b, c);
        store_c(ab, 32, mt * 16, 0, lane, c, false);
    }
    __syncthreads();
}

// Per-node MLP (H=32 fixed) on one LDS vector; lane j = hidden/output unit j.
__device__ inline void node_mlp(const float* in, int nin,
                                const float* W1, const float* b1,
                                const float* W2, const float* b2, int nout,
                                float* hid, float* out, int lane) {
    __syncthreads();
    {
        float s = b1[lane];
        for (int i = 0; i < nin; ++i) s += in[i] * W1[i * 32 + lane];
        hid[lane] = fmaxf(s, 0.f);
    }
    __syncthreads();
    if (lane < nout) {
        float s = b2[lane];
        for (int i = 0; i < 32; ++i) s += hid[i] * W2[i * nout + lane];
        out[lane] = s;
    }
    __syncthreads();
}

__device__ inline float angle3(float ax, float ay, float az, float bx, float by, float bz) {
    float cx = ay * bz - az * by, cy = az * bx - ax * bz, cz = ax * by - ay * bx;
    float cn = sqrtf(cx * cx + cy * cy + cz * cz);
    return atan2f(cn, ax * bx + ay * by + az * bz);
}

struct KParams {
    const float *pos, *oldw, *normals; const int* cols; const float* stddev;
    const _Float16* wts;
    const float *b1_l1, *b2_l1, *b1_l2, *b2_l2, *b1_l3, *b2_l3, *b1_l4, *b2_l4;
    const float *gW1a, *gb1a, *gW2a, *gb2a;   // lg
    const float *gW1b, *gb1b, *gW2b, *gb2b;   // lg2
    const float *gW1c, *gb1c, *gW2c, *gb2c;   // lg3
    float* out; int N;
};

__launch_bounds__(256, 1)
__global__ void gnn_sdf_fused(KParams P) {
    __shared__ __align__(16) _Float16 s_ab[8][32][32];   // per-wave activations (in/out)
    __shared__ __align__(16) _Float16 s_hb[8][32][64];   // per-wave hidden
    __shared__ float s_gin[8][20];
    __shared__ float s_ghid[8][32];
    __shared__ float s_gout[8][12];
    __shared__ float s_logit[8][32];

    const int tid  = threadIdx.x;
    const int lane = tid & 31;
    const int w    = tid >> 5;
    const int n    = blockIdx.x * 8 + w;
    const bool valid = (n < P.N);
    const int  nc  = valid ? n : (P.N - 1);
    const int  e   = nc * 32 + lane;

    _Float16* ab   = &s_ab[w][0][0];
    _Float16* hb   = &s_hb[w][0][0];
    _Float16* arow = ab + lane * 32;

    // ---- per-edge geometric features ----
    const float scale = 0.2f / P.stddev[0];
    const float prx = P.pos[nc * 3 + 0], pry = P.pos[nc * 3 + 1], prz = P.pos[nc * 3 + 2];
    const float nrx = P.normals[nc * 3 + 0], nry = P.normals[nc * 3 + 1], nrz = P.normals[nc * 3 + 2];
    const int   col = P.cols[e];
    const float cx  = (P.pos[col * 3 + 0] - prx) * scale;
    const float cy  = (P.pos[col * 3 + 1] - pry) * scale;
    const float cz  = (P.pos[col * 3 + 2] - prz) * scale;
    const float ncx = P.normals[col * 3 + 0], ncy = P.normals[col * 3 + 1], ncz = P.normals[col * 3 + 2];
    const float rr  = sqrtf(cx * cx + cy * cy + cz * cz);

    arow[0] = (_Float16)cx;
    arow[1] = (_Float16)cy;
    arow[2] = (_Float16)cz;
    arow[3] = (_Float16)P.oldw[e];
    arow[4] = (_Float16)rr;
    arow[5] = (_Float16)angle3(nrx, nry, nrz, cx, cy, cz);
    arow[6] = (_Float16)angle3(ncx, ncy, ncz, cx, cy, cz);
    arow[7] = (_Float16)angle3(nrx, nry, nrz, ncx, ncy, ncz);
#pragma unroll
    for (int i = 8; i < 32; ++i) arow[i] = (_Float16)0.f;

    // ---- l1 (8->32->16) ----
    edge_mlp(ab, hb, P.wts + OFF_L1_W1T, P.b1_l1, P.wts + OFF_L1_W2T, P.b2_l1, lane);

    // ---- lg: mean(x1) ++ normals -> 19 -> 32 -> 8 ----
    if (lane < 16) { float s = 0.f; for (int r = 0; r < 32; ++r) s += (float)ab[r * 32 + lane]; s_gin[w][lane] = s * (1.f / 32.f); }
    else if (lane < 19) s_gin[w][lane] = (lane == 16) ? nrx : (lane == 17) ? nry : nrz;
    node_mlp(s_gin[w], 19, P.gW1a, P.gb1a, P.gW2a, P.gb2a, 8, s_ghid[w], s_gout[w], lane);

#pragma unroll
    for (int j = 0; j < 8; ++j) arow[16 + j] = (_Float16)s_gout[w][j];
#pragma unroll
    for (int j = 24; j < 32; ++j) arow[j] = (_Float16)0.f;

    // ---- l2 (24->32->16) ----
    edge_mlp(ab, hb, P.wts + OFF_L2_W1T, P.b1_l2, P.wts + OFF_L2_W2T, P.b2_l2, lane);

    // ---- lg2: mean(x2) -> 16 -> 32 -> 8 ----
    if (lane < 16) { float s = 0.f; for (int r = 0; r < 32; ++r) s += (float)ab[r * 32 + lane]; s_gin[w][lane] = s * (1.f / 32.f); }
    node_mlp(s_gin[w], 16, P.gW1b, P.gb1b, P.gW2b, P.gb2b, 8, s_ghid[w], s_gout[w], lane);

#pragma unroll
    for (int j = 0; j < 8; ++j) arow[16 + j] = (_Float16)s_gout[w][j];
#pragma unroll
    for (int j = 24; j < 32; ++j) arow[j] = (_Float16)0.f;

    // ---- l3 (24->32->16) ----
    edge_mlp(ab, hb, P.wts + OFF_L3_W1T, P.b1_l3, P.wts + OFF_L3_W2T, P.b2_l3, lane);

    // ---- lg3: mean(x3) -> 16 -> 32 -> 12 ----
    if (lane < 16) { float s = 0.f; for (int r = 0; r < 32; ++r) s += (float)ab[r * 32 + lane]; s_gin[w][lane] = s * (1.f / 32.f); }
    node_mlp(s_gin[w], 16, P.gW1c, P.gb1c, P.gW2c, P.gb2c, 12, s_ghid[w], s_gout[w], lane);

    // ---- quaternion -> rotation, rotate cart ----
    {
        float q0 = s_gout[w][0], q1 = s_gout[w][1], q2 = s_gout[w][2], q3 = s_gout[w][3];
        float qn = sqrtf(q0 * q0 + q1 * q1 + q2 * q2 + q3 * q3) + 1e-8f;
        float qw = q0 / qn, qx = q1 / qn, qy = q2 / qn, qz = q3 / qn;
        float m00 = 1.f - 2.f * (qy * qy + qz * qz), m01 = 2.f * (qx * qy - qw * qz), m02 = 2.f * (qx * qz + qw * qy);
        float m10 = 2.f * (qx * qy + qw * qz), m11 = 1.f - 2.f * (qx * qx + qz * qz), m12 = 2.f * (qy * qz - qw * qx);
        float m20 = 2.f * (qx * qz - qw * qy), m21 = 2.f * (qy * qz + qw * qx), m22 = 1.f - 2.f * (qx * qx + qy * qy);
        float rcx = m00 * cx + m01 * cy + m02 * cz;
        float rcy = m10 * cx + m11 * cy + m12 * cz;
        float rcz = m20 * cx + m21 * cy + m22 * cz;
#pragma unroll
        for (int j = 0; j < 8; ++j) arow[16 + j] = (_Float16)s_gout[w][4 + j];
        arow[24] = (_Float16)rcx; arow[25] = (_Float16)rcy; arow[26] = (_Float16)rcz;
#pragma unroll
        for (int j = 27; j < 32; ++j) arow[j] = (_Float16)0.f;
    }
    __syncthreads();

    // ---- l4: 27(->32) -> 64 relu -> 1 ----
    {
        const _Float16* w1T = P.wts + OFF_L4_W1T;
#pragma unroll
        for (int mt = 0; mt < 2; ++mt) {
            v16h a = afrag(ab, 32, mt * 16, lane, 0);
#pragma unroll
            for (int nt = 0; nt < 4; ++nt) {
                v16h b = bfrag(w1T, 32, nt * 16, lane, 0);
                v8f  c = cinit(P.b1_l4, nt * 16, lane);
                c = WMMA(a, b, c);
                store_c(hb, 64, mt * 16, nt * 16, lane, c, true);
            }
        }
        __syncthreads();
        const _Float16* w2T = P.wts + OFF_L4_W2T;
#pragma unroll
        for (int mt = 0; mt < 2; ++mt) {
            float bv = ((lane & 15) == 0) ? P.b2_l4[0] : 0.f;
            v8f c;
#pragma unroll
            for (int r = 0; r < 8; ++r) c[r] = bv;
            v16h a0 = afrag(hb, 64, mt * 16, lane, 0);
            v16h b0 = bfrag(w2T, 64, 0, lane, 0);
            c = WMMA(a0, b0, c);
            v16h a1 = afrag(hb, 64, mt * 16, lane, 32);
            v16h b1 = bfrag(w2T, 64, 0, lane, 32);
            c = WMMA(a1, b1, c);
            if ((lane & 15) == 0) {
                int mb = mt * 16 + ((lane >> 4) << 3);
#pragma unroll
                for (int r = 0; r < 8; ++r) s_logit[w][mb + r] = c[r];
            }
        }
        __syncthreads();
    }

    // ---- softmax over this node's 32 edges ----
    {
        float lg = s_logit[w][lane];
        float mx = lg;
#pragma unroll
        for (int d = 16; d >= 1; d >>= 1) mx = fmaxf(mx, __shfl_xor(mx, d, 32));
        float ex = __expf(lg - mx);
        float sm = ex;
#pragma unroll
        for (int d = 16; d >= 1; d >>= 1) sm += __shfl_xor(sm, d, 32);
        if (valid) P.out[n * 32 + lane] = ex / sm;
    }
}

extern "C" void kernel_launch(void* const* d_in, const int* in_sizes, int n_in,
                              void* d_out, int out_size, void* d_ws, size_t ws_size,
                              hipStream_t stream) {
    const int E = in_sizes[3] / 2;
    const int N = in_sizes[0] / 3;
    _Float16* wts = (_Float16*)d_ws;

    prep_weights<<<(WTS_TOTAL + 255) / 256, 256, 0, stream>>>(
        (const float*)d_in[6],  (const float*)d_in[8],    // l1 W1, W2
        (const float*)d_in[14], (const float*)d_in[16],   // l2
        (const float*)d_in[22], (const float*)d_in[24],   // l3
        (const float*)d_in[30], (const float*)d_in[32],   // l4
        wts);

    KParams P;
    P.pos     = (const float*)d_in[0];
    P.oldw    = (const float*)d_in[1];
    P.normals = (const float*)d_in[2];
    P.cols    = ((const int*)d_in[3]) + E;
    P.stddev  = (const float*)d_in[5];
    P.wts     = wts;
    P.b1_l1 = (const float*)d_in[7];  P.b2_l1 = (const float*)d_in[9];
    P.b1_l2 = (const float*)d_in[15]; P.b2_l2 = (const float*)d_in[17];
    P.b1_l3 = (const float*)d_in[23]; P.b2_l3 = (const float*)d_in[25];
    P.b1_l4 = (const float*)d_in[31]; P.b2_l4 = (const float*)d_in[33];
    P.gW1a = (const float*)d_in[10]; P.gb1a = (const float*)d_in[11];
    P.gW2a = (const float*)d_in[12]; P.gb2a = (const float*)d_in[13];
    P.gW1b = (const float*)d_in[18]; P.gb1b = (const float*)d_in[19];
    P.gW2b = (const float*)d_in[20]; P.gb2b = (const float*)d_in[21];
    P.gW1c = (const float*)d_in[26]; P.gb1c = (const float*)d_in[27];
    P.gW2c = (const float*)d_in[28]; P.gb2c = (const float*)d_in[29];
    P.out = (float*)d_out;
    P.N   = N;

    gnn_sdf_fused<<<(N + 7) / 8, 256, 0, stream>>>(P);
}
